// GRU_72688026517903
// MI455X (gfx1250) — compile-verified, matrix operand order checked by reading, not run on hardware
//
#include <hip/hip_runtime.h>
#include <hip/hip_bf16.h>

typedef __attribute__((ext_vector_type(16))) __bf16 v16bf;
typedef __attribute__((ext_vector_type(8)))  __bf16 v8bf;
typedef __attribute__((ext_vector_type(8)))  float  v8f;

#define T_ 512
#define B_ 256
#define D_ 85
#define H_ 512
#define O_ 33
#define BT 16          // batch rows per block
#define DP 96          // D padded to multiple of 32
#define XP 104         // padded LDS row for x tiles (bf16)
#define HP 520         // padded LDS row for h tiles (bf16) -> breaks bank conflicts
#define KT_H (H_/32)   // 16 k-tiles over hidden
#define KT_X (DP/32)   // 3 k-tiles over input
#define NT_H (H_/16)   // 32 n-tiles over hidden
#define NT_O 3         // 48 cols, 33 valid

// workspace layout (bf16 elements): pre-swizzled WMMA B-fragments
#define WH_ELEMS (3*KT_H*NT_H*512)   // 786432
#define WX_ELEMS (3*KT_X*NT_H*512)   // 147456
#define WO_ELEMS (KT_H*NT_O*512)     //  24576
#define TOTAL_FRAGS (3*KT_H*NT_H + 3*KT_X*NT_H + KT_H*NT_O) // 1872

// ---------------------------------------------------------------------------
// Prep: convert fp32 weights -> bf16 B-matrix fragments in per-lane order.
//   j = v*2+p :  k = kt*32 + (v>=4?16:0) + (l>>4)*8 + (v&3)*2 + p,  n = nt*16 + (l&15)
// ---------------------------------------------------------------------------
__global__ void gru_prep(const float* __restrict__ Whr, const float* __restrict__ Whz,
                         const float* __restrict__ Whh,
                         const float* __restrict__ Wxr, const float* __restrict__ Wxz,
                         const float* __restrict__ Wxh,
                         const float* __restrict__ Wro, __bf16* __restrict__ ws)
{
    int tid = blockIdx.x * blockDim.x + threadIdx.x;
    if (tid >= TOTAL_FRAGS * 32) return;
    int frag = tid >> 5, lane = tid & 31;
    int nl = lane & 15, khalf = lane >> 4;
    v16bf o;
    if (frag < 3*KT_H*NT_H) {                              // W_h{r,z,h}: B(k,n) = W[n][k]
        int g  = frag / (KT_H*NT_H);
        int kt = (frag / NT_H) % KT_H;
        int nt = frag % NT_H;
        const float* W = (g==0) ? Whr : (g==1) ? Whz : Whh;
        int n = nt*16 + nl;
        #pragma unroll
        for (int j = 0; j < 16; ++j) {
            int v = j >> 1, p = j & 1;
            int k = kt*32 + ((v >= 4) ? 16 : 0) + khalf*8 + (v & 3)*2 + p;
            o[j] = (__bf16)W[(size_t)n*H_ + k];
        }
    } else if (frag < 3*KT_H*NT_H + 3*KT_X*NT_H) {         // W_x{r,z,h}: zero-padded K
        int f  = frag - 3*KT_H*NT_H;
        int g  = f / (KT_X*NT_H);
        int kt = (f / NT_H) % KT_X;
        int nt = f % NT_H;
        const float* W = (g==0) ? Wxr : (g==1) ? Wxz : Wxh;
        int n = nt*16 + nl;
        #pragma unroll
        for (int j = 0; j < 16; ++j) {
            int v = j >> 1, p = j & 1;
            int k = kt*32 + ((v >= 4) ? 16 : 0) + khalf*8 + (v & 3)*2 + p;
            o[j] = (k < D_) ? (__bf16)W[(size_t)n*D_ + k] : (__bf16)0.0f;
        }
    } else {                                               // W_ro: zero-padded N
        int f  = frag - (3*KT_H*NT_H + 3*KT_X*NT_H);
        int kt = f / NT_O;
        int nt = f % NT_O;
        int n = nt*16 + nl;
        #pragma unroll
        for (int j = 0; j < 16; ++j) {
            int v = j >> 1, p = j & 1;
            int k = kt*32 + ((v >= 4) ? 16 : 0) + khalf*8 + (v & 3)*2 + p;
            o[j] = (n < O_) ? (__bf16)Wro[(size_t)n*H_ + k] : (__bf16)0.0f;
        }
    }
    *(v16bf*)(ws + (size_t)frag*512 + lane*16) = o;
}

// ---------------------------------------------------------------------------
// Recurrent scan kernel
// ---------------------------------------------------------------------------
__device__ __forceinline__ v8f wmma_bf16(v16bf a, v16bf b, v8f c) {
    return __builtin_amdgcn_wmma_f32_16x16x32_bf16(false, a, false, b, (short)0, c,
                                                   false, false);
}

union AFrag { struct { v8bf lo, hi; } h2; v16bf v; };

// A operand from LDS: 16-bit A layout -> two 16B ds_load_b128 per lane
__device__ __forceinline__ v16bf load_a(const __bf16* row, int k0) {
    AFrag f;
    f.h2.lo = *(const v8bf*)(row + k0);
    f.h2.hi = *(const v8bf*)(row + k0 + 16);
    return f.v;
}
// B operand from global: lane-major pre-swizzled fragment, 32B per lane
__device__ __forceinline__ v16bf load_b(const __bf16* frag, int lane) {
    return *(const v16bf*)(frag + lane*16);
}

__device__ __forceinline__ float fast_sigmoid(float xx) {
    float e = __builtin_amdgcn_exp2f(-1.4426950408889634f * xx);
    return __builtin_amdgcn_rcpf(1.0f + e);
}
__device__ __forceinline__ float fast_tanh(float xx) {
    float e = __builtin_amdgcn_exp2f(-2.8853900817779268f * xx);
    return (1.0f - e) * __builtin_amdgcn_rcpf(1.0f + e);
}

__global__ __launch_bounds__(512, 1) void gru_scan(
    const float* __restrict__ x,  const float* __restrict__ rn,
    const float* __restrict__ zn, const float* __restrict__ hn,
    const float* __restrict__ bhr, const float* __restrict__ bhz,
    const float* __restrict__ bhh, const float* __restrict__ bro,
    const __bf16* whF, const __bf16* wxF,
    const __bf16* woF, float* __restrict__ out)
{
    __shared__ __align__(16) __bf16 hS [BT][HP];   // current hidden state (bf16)
    __shared__ __align__(16) __bf16 rhS[BT][HP];   // R * h
    __shared__ __align__(16) __bf16 xS [3][BT][XP];// (x+noise_g) staged per step

    const int tid   = threadIdx.x;
    const int lane  = tid & 31;
    const int wave  = tid >> 5;
    const int nl    = lane & 15;
    const int khalf = lane >> 4;
    const int b0    = blockIdx.x * BT;
    const int n0    = wave * 32;        // this wave's 32 output columns
    const int ntg0  = n0 >> 4;          // global 16-col tile index

    for (int i = tid; i < BT*HP; i += 512) (&hS[0][0])[i] = (__bf16)0.0f;

    const float bR[2] = { bhr[n0+nl], bhr[n0+16+nl] };
    const float bZ[2] = { bhz[n0+nl], bhz[n0+16+nl] };
    const float bHh[2]= { bhh[n0+nl], bhh[n0+16+nl] };
    const float bO    = (wave < NT_O && (wave*16+nl) < O_) ? bro[wave*16+nl] : 0.0f;

    __syncthreads();

    for (int t = 0; t < T_; ++t) {
        // Optimization barrier: weight-fragment loads below must NOT be hoisted
        // out of the t-loop (they are 1.9MB total and L2-resident; hoisting them
        // into registers spills to scratch).
        asm volatile("" ::: "memory");

        // ---- stage (x + gate noise) tiles into LDS as bf16, K padded to 96 ----
        const size_t rowBase = ((size_t)t * B_ + b0) * D_;
        for (int i = tid; i < 3*BT*DP; i += 512) {     // 4608 = 9 full passes
            int g = i / (BT*DP);
            int m = (i / DP) % BT;
            int k = i % DP;
            float v = 0.0f;
            if (k < D_) {
                const float* nz = (g==0) ? rn : (g==1) ? zn : hn;
                v = x[rowBase + (size_t)m*D_ + k] + nz[rowBase + (size_t)m*D_ + k];
            }
            xS[g][m][k] = (__bf16)v;
        }
        if (t + 1 < T_ && tid < BT)                     // pull next step's x into caches
            __builtin_prefetch(&x[rowBase + (size_t)B_*D_ + (size_t)tid*D_], 0, 1);
        __syncthreads();

        // ---- phase 1: R, Z gates (GEMM over x-part then h-part) ----
        v8f aR0 = {}, aR1 = {}, aZ0 = {}, aZ1 = {};
        #pragma unroll 1
        for (int kt = 0; kt < KT_X; ++kt) {
            const int k0 = kt*32 + khalf*8;
            v16bf ar = load_a(&xS[0][nl][0], k0);
            v16bf az = load_a(&xS[1][nl][0], k0);
            aR0 = wmma_bf16(ar, load_b(wxF + ((size_t)((0*KT_X+kt)*NT_H + ntg0    ))*512, lane), aR0);
            aR1 = wmma_bf16(ar, load_b(wxF + ((size_t)((0*KT_X+kt)*NT_H + ntg0 + 1))*512, lane), aR1);
            aZ0 = wmma_bf16(az, load_b(wxF + ((size_t)((1*KT_X+kt)*NT_H + ntg0    ))*512, lane), aZ0);
            aZ1 = wmma_bf16(az, load_b(wxF + ((size_t)((1*KT_X+kt)*NT_H + ntg0 + 1))*512, lane), aZ1);
        }
        #pragma unroll 2
        for (int kt = 0; kt < KT_H; ++kt) {
            const int k0 = kt*32 + khalf*8;
            v16bf ah = load_a(&hS[nl][0], k0);
            aR0 = wmma_bf16(ah, load_b(whF + ((size_t)((0*KT_H+kt)*NT_H + ntg0    ))*512, lane), aR0);
            aR1 = wmma_bf16(ah, load_b(whF + ((size_t)((0*KT_H+kt)*NT_H + ntg0 + 1))*512, lane), aR1);
            aZ0 = wmma_bf16(ah, load_b(whF + ((size_t)((1*KT_H+kt)*NT_H + ntg0    ))*512, lane), aZ0);
            aZ1 = wmma_bf16(ah, load_b(whF + ((size_t)((1*KT_H+kt)*NT_H + ntg0 + 1))*512, lane), aZ1);
        }

        float zsv[2][8];
        #pragma unroll
        for (int nt = 0; nt < 2; ++nt) {
            #pragma unroll
            for (int i = 0; i < 8; ++i) {
                float rv = fast_sigmoid(((nt==0) ? aR0[i] : aR1[i]) + bR[nt]);
                float zv = fast_sigmoid(((nt==0) ? aZ0[i] : aZ1[i]) + bZ[nt]);
                int m = i + khalf*8;
                int n = n0 + nt*16 + nl;
                float hv = (float)hS[m][n];
                rhS[m][n] = (__bf16)(rv * hv);
                zsv[nt][i] = zv;
            }
        }
        __syncthreads();

        // ---- phase 2: candidate H_hat and state update ----
        v8f aH0 = {}, aH1 = {};
        #pragma unroll 1
        for (int kt = 0; kt < KT_X; ++kt) {
            const int k0 = kt*32 + khalf*8;
            v16bf ax = load_a(&xS[2][nl][0], k0);
            aH0 = wmma_bf16(ax, load_b(wxF + ((size_t)((2*KT_X+kt)*NT_H + ntg0    ))*512, lane), aH0);
            aH1 = wmma_bf16(ax, load_b(wxF + ((size_t)((2*KT_X+kt)*NT_H + ntg0 + 1))*512, lane), aH1);
        }
        #pragma unroll 2
        for (int kt = 0; kt < KT_H; ++kt) {
            const int k0 = kt*32 + khalf*8;
            v16bf ar = load_a(&rhS[nl][0], k0);
            aH0 = wmma_bf16(ar, load_b(whF + ((size_t)((2*KT_H+kt)*NT_H + ntg0    ))*512, lane), aH0);
            aH1 = wmma_bf16(ar, load_b(whF + ((size_t)((2*KT_H+kt)*NT_H + ntg0 + 1))*512, lane), aH1);
        }
        #pragma unroll
        for (int nt = 0; nt < 2; ++nt) {
            #pragma unroll
            for (int i = 0; i < 8; ++i) {
                float hh = fast_tanh(((nt==0) ? aH0[i] : aH1[i]) + bHh[nt]);
                int m = i + khalf*8;
                int n = n0 + nt*16 + nl;
                float hv = (float)hS[m][n];
                float z  = zsv[nt][i];
                hS[m][n] = (__bf16)(z*hv + (1.0f - z)*hh);
            }
        }
        __syncthreads();   // h_t complete and visible

        // ---- fused output projection: out[t] = h_t @ W_ro^T + b_ro ----
        if (wave < NT_O) {
            v8f aO = {};
            #pragma unroll 2
            for (int kt = 0; kt < KT_H; ++kt) {
                const int k0 = kt*32 + khalf*8;
                v16bf ah = load_a(&hS[nl][0], k0);
                aO = wmma_bf16(ah, load_b(woF + ((size_t)(kt*NT_O + wave))*512, lane), aO);
            }
            int n = wave*16 + nl;
            if (n < O_) {
                #pragma unroll
                for (int i = 0; i < 8; ++i) {
                    int m = i + khalf*8;
                    out[((size_t)t*B_ + b0 + m)*O_ + n] = aO[i] + bO;
                }
            }
        }
        // waves 3..15 proceed to next step's staging; hS is untouched until the
        // next phase-2, which sits behind two barriers -> no hazard with out-GEMM.
    }
}

// ---------------------------------------------------------------------------
extern "C" void kernel_launch(void* const* d_in, const int* in_sizes, int n_in,
                              void* d_out, int out_size, void* d_ws, size_t ws_size,
                              hipStream_t stream)
{
    const float* x   = (const float*)d_in[0];
    const float* rn  = (const float*)d_in[1];
    const float* zn  = (const float*)d_in[2];
    const float* hn  = (const float*)d_in[3];
    const float* Wxr = (const float*)d_in[4];
    const float* Wxz = (const float*)d_in[5];
    const float* Wxh = (const float*)d_in[6];
    const float* Whr = (const float*)d_in[7];
    const float* bhr = (const float*)d_in[8];
    const float* Whz = (const float*)d_in[9];
    const float* bhz = (const float*)d_in[10];
    const float* Whh = (const float*)d_in[11];
    const float* bhh = (const float*)d_in[12];
    const float* Wro = (const float*)d_in[13];
    const float* bro = (const float*)d_in[14];
    float* out = (float*)d_out;

    __bf16* whF = (__bf16*)d_ws;                                    // ~1.9 MB used
    __bf16* wxF = (__bf16*)((char*)d_ws + (size_t)WH_ELEMS * 2);
    __bf16* woF = (__bf16*)((char*)d_ws + (size_t)(WH_ELEMS + WX_ELEMS) * 2);

    const int prepThreads = TOTAL_FRAGS * 32;                       // 59904
    gru_prep<<<(prepThreads + 255) / 256, 256, 0, stream>>>(
        Whr, Whz, Whh, Wxr, Wxz, Wxh, Wro, whF);

    gru_scan<<<B_ / BT, 512, 0, stream>>>(
        x, rn, zn, hn, bhr, bhz, bhh, bro, whF, wxF, woF, out);
}
